// MicroTopoBrain_3934190043715
// MI455X (gfx1250) — matrix operand order repaired and checked by reading, not compiled, not run on hardware
//
#include <hip/hip_runtime.h>
#include <hip/hip_bf16.h>
#include <math.h>

typedef __attribute__((ext_vector_type(16))) _Float16 v16h;
typedef __attribute__((ext_vector_type(8)))  float    v8f;
typedef __attribute__((__vector_size__(16))) int      i32x4_;

#define BB    32768
#define FIN   64
#define NN    64
#define DD    16
#define HID   1024
#define NC    10
#define NPROJ 4

// scalar slots in workspace
#define S_SUMX     0
#define S_SUMSQX   1
#define S_SUMABS_E 2
#define S_SUM_XF   3
#define S_SUMSQ_XF 4
#define S_SUMABS_V 5
#define S_CTRL0    6
#define S_CTRL2    7
#define S_SUM_FL   8
#define S_SUMSQ_FL 9
#define S_GAIN     10
#define S_ORTHO    11
#define S_ENT_LAST 12
#define S_NORM_WS  13

// ---- CDNA5 async global->LDS copy (ASYNCcnt path), with safe fallback ----
__device__ __forceinline__ void async_copy_b128(void* lds_dst, const void* g_src)
{
#if defined(__has_builtin) && __has_builtin(__builtin_amdgcn_global_load_async_to_lds_b128)
    __builtin_amdgcn_global_load_async_to_lds_b128(
        (__attribute__((address_space(1))) i32x4_*)(g_src),
        (__attribute__((address_space(3))) i32x4_*)(lds_dst),
        0, 0);
#else
    *(float4*)lds_dst = *(const float4*)g_src;
#endif
}

__device__ __forceinline__ void async_wait0()
{
#if defined(__has_builtin) && __has_builtin(__builtin_amdgcn_s_wait_asynccnt)
    __builtin_amdgcn_s_wait_asynccnt(0);
#else
    asm volatile("s_wait_asynccnt 0x0" ::: "memory");
#endif
}

__device__ __forceinline__ float sigmoidf_(float x) { return 1.0f / (1.0f + expf(-x)); }

// Linear(in->16) -> LayerNorm -> tanh(*g+be) -> Linear(16->3) -> sigmoid
__device__ void regulator3(const float* sig, int in_dim,
                           const float* w1, const float* b1,
                           const float* g,  const float* be,
                           const float* w2, const float* b2,
                           float* out)
{
    float h[16];
    float mu = 0.f;
    for (int i = 0; i < 16; ++i) {
        float s = b1[i];
        for (int j = 0; j < in_dim; ++j) s += w1[i * in_dim + j] * sig[j];
        h[i] = s; mu += s;
    }
    mu *= (1.f / 16.f);
    float var = 0.f;
    for (int i = 0; i < 16; ++i) { float d = h[i] - mu; var += d * d; }
    var *= (1.f / 16.f);
    float inv = rsqrtf(var + 1e-5f);
    for (int i = 0; i < 16; ++i) h[i] = tanhf((h[i] - mu) * inv * g[i] + be[i]);
    for (int o = 0; o < 3; ++o) {
        float s = b2[o];
        for (int i = 0; i < 16; ++i) s += w2[o * 16 + i] * h[i];
        out[o] = sigmoidf_(s);
    }
}

// ---------------- K0: zero scratch scalars ----------------
__global__ void k_zero(float* S)
{
    for (int i = threadIdx.x; i < 1024; i += blockDim.x) S[i] = 0.f;
}

// ---------------- K1a: var(x) sums ----------------
__global__ void __launch_bounds__(256) k_xstats(const float* __restrict__ x, float* S, int n)
{
    float s = 0.f, q = 0.f;
    for (int i = blockIdx.x * blockDim.x + threadIdx.x; i < n; i += gridDim.x * blockDim.x) {
        float v = x[i]; s += v; q += v * v;
    }
    __shared__ float rs[256], rq[256];
    rs[threadIdx.x] = s; rq[threadIdx.x] = q; __syncthreads();
    for (int st = 128; st > 0; st >>= 1) {
        if ((int)threadIdx.x < st) { rs[threadIdx.x] += rs[threadIdx.x + st]; rq[threadIdx.x] += rq[threadIdx.x + st]; }
        __syncthreads();
    }
    if (threadIdx.x == 0) { atomicAdd(&S[S_SUMX], rs[0]); atomicAdd(&S[S_SUMSQX], rq[0]); }
}

// ---------------- K1: embed GEMM (WMMA f16->f32), + sum|x_embed| ----------------
// out[b,h] = sum_k x[b,k]*We[h,k] + be[h]
// Each wave: one 16-row batch tile, A fragment kept resident, swept over 4 HID tiles.
__global__ void __launch_bounds__(256) k_embed(const float* __restrict__ x,
                                               const float* __restrict__ We,
                                               const float* __restrict__ be_,
                                               float* __restrict__ out, float* S)
{
    int lane = threadIdx.x & 31;
    int wave = threadIdx.x >> 5;
    int job  = blockIdx.x * 8 + wave;        // 32768 jobs
    int tb   = job >> 4;                     // batch tile 0..2047
    int thg  = job & 15;                     // hid tile group 0..15 (4 tiles each)
    int row  = lane & 15;
    int half = lane >> 4;
    const float* arow = x + (size_t)(tb * 16 + row) * FIN;

    // prefetch upcoming weight rows into cache
    for (int u = 0; u < 4; ++u)
        __builtin_prefetch(We + (size_t)((thg * 4 + u) * 16 + row) * FIN, 0, 3);

    // load + convert A fragment once (K = 64 -> two 16x32 fragments)
    v16h a[2];
    for (int s = 0; s < 2; ++s) {
        int ka = s * 32 + half * 16;
        for (int t = 0; t < 8; ++t) {
            a[s][t]     = (_Float16)arow[ka + t];
            a[s][t + 8] = (_Float16)arow[ka + 8 + t];
        }
    }

    int col = row;
    float sabs = 0.f;
    for (int u = 0; u < 4; ++u) {
        int th = thg * 4 + u;
        const float* brow = We + (size_t)(th * 16 + row) * FIN;
        v8f c = {0.f, 0.f, 0.f, 0.f, 0.f, 0.f, 0.f, 0.f};
        for (int s = 0; s < 2; ++s) {
            int kb = s * 32 + half * 16;
            v16h b;
            for (int t = 0; t < 16; ++t) b[t] = (_Float16)brow[kb + t];
            c = __builtin_amdgcn_wmma_f32_16x16x32_f16(false, a[s], false, b, (short)0, c, false, false);
        }
        float bias = be_[th * 16 + col];
        int rowoff = half * 8;
        for (int r = 0; r < 8; ++r) {
            float v = c[r] + bias;
            out[(size_t)(tb * 16 + rowoff + r) * HID + th * 16 + col] = v;
            sabs += fabsf(v);
        }
    }
    for (int off = 16; off > 0; off >>= 1) sabs += __shfl_down(sabs, off, 32);
    if (lane == 0) atomicAdd(&S[S_SUMABS_E], sabs);
}

// ---------------- K2: norms, ortho, pr-regulator, adjacency ----------------
__global__ void __launch_bounds__(256) k_prep(const float* __restrict__ adjw,
                                              const float* __restrict__ mask,
                                              const float* __restrict__ Wslow,
                                              const float* __restrict__ basis,
                                              const float* pw1, const float* pb1, const float* pg,
                                              const float* pbe, const float* pw2, const float* pb2,
                                              float* S, float* adjm, float* dout_ortho)
{
    __shared__ float red[256];
    __shared__ float atmp[4096];
    __shared__ float rinv[64];
    __shared__ float bc[4];
    int t = threadIdx.x;

    float s = 0.f;
    for (int i = t; i < 4096; i += 256) { float v = adjw[i]; s += v * v; }
    red[t] = s; __syncthreads();
    for (int st = 128; st > 0; st >>= 1) { if (t < st) red[t] += red[t + st]; __syncthreads(); }
    float norm_adj = sqrtf(red[0]);
    __syncthreads();

    s = 0.f;
    for (int i = t; i < 256; i += 256) { float v = Wslow[i]; s += v * v; }
    red[t] = s; __syncthreads();
    for (int st = 128; st > 0; st >>= 1) { if (t < st) red[t] += red[t + st]; __syncthreads(); }
    if (t == 0) S[S_NORM_WS] = sqrtf(red[0]);

    if (t == 0) {
        float o = 0.f;
        for (int i = 0; i < 2; ++i)
            for (int j = 0; j < 2; ++j) {
                float gg = 0.f;
                for (int d = 0; d < 16; ++d) gg += basis[i * 16 + d] * basis[j * 16 + d];
                gg -= (i == j) ? 1.f : 0.f;
                o += gg * gg;
            }
        o = sqrtf(o);
        S[S_ORTHO] = o; *dout_ortho = o;

        float Mx = (float)BB * (float)FIN;
        float varx  = (S[S_SUMSQX] - S[S_SUMX] * S[S_SUMX] / Mx) / (Mx - 1.f);
        float meane = S[S_SUMABS_E] / ((float)BB * (float)HID);
        float sig[3] = {varx, meane, norm_adj};
        float outp[3];
        regulator3(sig, 3, pw1, pb1, pg, pbe, pw2, pb2, outp);
        bc[0] = outp[0];
    }
    __syncthreads();
    float pl0 = bc[0];
    for (int i = t; i < 4096; i += 256) atmp[i] = sigmoidf_(adjw[i] * pl0) * mask[i];
    __syncthreads();
    if (t < 64) {
        float r = 0.f;
        for (int m = 0; m < 64; ++m) r += atmp[t * 64 + m];
        rinv[t] = 1.f / fmaxf(r, 1e-6f);
    }
    __syncthreads();
    for (int i = t; i < 4096; i += 256) adjm[i] = atmp[i] * rinv[i >> 6];
}

// ---------------- K3: x_agg = adj @ x_embed (per batch row) + xf/v stats ----------------
__global__ void __launch_bounds__(256) k_agg(const float* __restrict__ xe,
                                             const float* __restrict__ adjm,
                                             const float* __restrict__ Vs,
                                             float* __restrict__ xa, float* S)
{
    __shared__ __align__(16) float adj_s[4096];
    __shared__ __align__(16) float in_s[8][1024];
    __shared__ float vs_s[256];
    __shared__ float red[256];
    int t = threadIdx.x, lane = t & 31, wave = t >> 5;
    // async stage adjacency into LDS
    for (int i = t; i < 1024; i += 256)
        async_copy_b128(&((float4*)adj_s)[i], &((const float4*)adjm)[i]);
    for (int i = t; i < 256; i += 256) vs_s[i] = Vs[i];

    float acc_s = 0.f, acc_q = 0.f, acc_av = 0.f;
    for (int g = blockIdx.x; g < BB / 8; g += gridDim.x) {
        __syncthreads();
        const float4* src = (const float4*)(xe + (size_t)g * 8 * HID);
        for (int i = t; i < 2048; i += 256)
            async_copy_b128(&((float4*)in_s)[i], &src[i]);
        async_wait0();
        __syncthreads();
        float* row = in_s[wave];
        float o[32];
        for (int tt = 0; tt < 32; ++tt) {
            int j = tt * 32 + lane;
            int n = j >> 4, d = j & 15;
            float sum = 0.f;
            const float* ar = adj_s + n * 64;
            for (int m = 0; m < 64; ++m) sum += ar[m] * row[m * 16 + d];
            o[tt] = sum;
        }
        float* dst = xa + (size_t)(g * 8 + wave) * HID;
        for (int tt = 0; tt < 32; ++tt) { int j = tt * 32 + lane; dst[j] = o[tt]; }
        for (int tt = 0; tt < 32; ++tt) { int j = tt * 32 + lane; row[j] = o[tt]; }  // wave-local reuse
        for (int nn = 0; nn < 2; ++nn) {
            int n = lane + nn * 32;
            float xv[16];
            for (int d = 0; d < 16; ++d) { float v = row[n * 16 + d]; xv[d] = v; acc_s += v; acc_q += v * v; }
            for (int dp = 0; dp < 16; ++dp) {
                float v = 0.f;
                for (int d = 0; d < 16; ++d) v += vs_s[dp * 16 + d] * xv[d];
                acc_av += fabsf(v);
            }
        }
    }
    __syncthreads();
    red[t] = acc_s; __syncthreads();
    for (int st = 128; st > 0; st >>= 1) { if (t < st) red[t] += red[t + st]; __syncthreads(); }
    if (t == 0) atomicAdd(&S[S_SUM_XF], red[0]);
    __syncthreads(); red[t] = acc_q; __syncthreads();
    for (int st = 128; st > 0; st >>= 1) { if (t < st) red[t] += red[t + st]; __syncthreads(); }
    if (t == 0) atomicAdd(&S[S_SUMSQ_XF], red[0]);
    __syncthreads(); red[t] = acc_av; __syncthreads();
    for (int st = 128; st > 0; st >>= 1) { if (t < st) red[t] += red[t + st]; __syncthreads(); }
    if (t == 0) atomicAdd(&S[S_SUMABS_V], red[0]);
}

// ---------------- K4: cr-regulator (1 thread) ----------------
__global__ void k_cr(const float* w1, const float* b1, const float* g, const float* be,
                     const float* w2, const float* b2, float* S)
{
    float M = (float)BB * (float)HID;
    float varxf = (S[S_SUMSQ_XF] - S[S_SUM_XF] * S[S_SUM_XF] / M) / (M - 1.f);
    float mav = S[S_SUMABS_V] / M;
    float sig[3] = {varxf, mav, S[S_NORM_WS]};
    float out[3];
    regulator3(sig, 3, w1, b1, g, be, w2, b2, out);
    S[S_CTRL0] = out[0]; S[S_CTRL2] = out[2];
}

// ---------------- K5: continuum + per-node pre-stats ----------------
__global__ void __launch_bounds__(256) k_cont(const float* __restrict__ xa,
                                              const float* __restrict__ Vs,
                                              const float* __restrict__ Wf,
                                              const float* __restrict__ Wq,
                                              const float* __restrict__ gw,
                                              const float* __restrict__ gb,
                                              const float* __restrict__ basis,
                                              float* S, float* nodeAcc,
                                              float* __restrict__ xp)
{
    __shared__ __align__(16) float in_s[8][1024];
    __shared__ float vs_s[256], wf_s[256], wq_s[256];
    __shared__ float nacc[3][64];
    int t = threadIdx.x, lane = t & 31, wave = t >> 5;
    for (int i = t; i < 256; i += 256) { vs_s[i] = Vs[i]; wf_s[i] = Wf[i]; wq_s[i] = Wq[i]; }
    for (int i = t; i < 192; i += 256) ((float*)nacc)[i] = 0.f;
    float ctrl0 = S[S_CTRL0], ctrl2 = S[S_CTRL2];
    float gwv[16]; for (int d = 0; d < 16; ++d) gwv[d] = gw[d];
    float gbv = gb[0];
    float b0[16], b1v[16];
    for (int d = 0; d < 16; ++d) { b0[d] = basis[d]; b1v[d] = basis[16 + d]; }
    const float scale = 4.0f + 1e-8f;   // sqrt(D)+EPS
    for (int g = blockIdx.x; g < BB / 8; g += gridDim.x) {
        __syncthreads();
        const float4* src = (const float4*)(xa + (size_t)g * 8 * HID);
        for (int i = t; i < 2048; i += 256)
            async_copy_b128(&((float4*)in_s)[i], &src[i]);
        async_wait0();
        __syncthreads();
        float* row = in_s[wave];
        float* dst = xp + (size_t)(g * 8 + wave) * HID;
        for (int nn = 0; nn < 2; ++nn) {
            int n = lane + nn * 32;
            float xv[16], vv[16], fv[16], p[16];
            for (int d = 0; d < 16; ++d) xv[d] = row[n * 16 + d];
            float gdot = gbv;
            for (int dp = 0; dp < 16; ++dp) {
                float v = 0.f, f = 0.f;
                for (int d = 0; d < 16; ++d) { v += vs_s[dp * 16 + d] * xv[d]; f += wf_s[dp * 16 + d] * xv[d]; }
                vv[dp] = v; fv[dp] = f; gdot += gwv[dp] * v;
            }
            float gate = sigmoidf_(gdot) * ctrl0;
            float ssum = 0.f, ssq = 0.f;
            for (int d = 0; d < 16; ++d) {
                float pv = gate * (vv[d] + fv[d] * ctrl2);
                p[d] = pv; dst[n * 16 + d] = pv;
                ssum += pv; ssq += pv * pv;
            }
            float q[16];
            for (int dp = 0; dp < 16; ++dp) {
                float v = 0.f;
                for (int d = 0; d < 16; ++d) v += wq_s[dp * 16 + d] * p[d];
                q[dp] = v;
            }
            float l0 = 0.f, l1 = 0.f;
            for (int d = 0; d < 16; ++d) { l0 += b0[d] * q[d]; l1 += b1v[d] * q[d]; }
            l0 /= scale; l1 /= scale;
            float m = fmaxf(l0, l1);
            float e0 = expf(l0 - m), e1 = expf(l1 - m);
            float inv = 1.f / (e0 + e1);
            float w0 = e0 * inv, w1 = e1 * inv;
            float ent = -(w0 * logf(w0 + 1e-8f) + w1 * logf(w1 + 1e-8f));
            atomicAdd(&nacc[0][n], ssum);
            atomicAdd(&nacc[1][n], ssq);
            atomicAdd(&nacc[2][n], ent);
        }
    }
    __syncthreads();
    for (int i = t; i < 192; i += 256) atomicAdd(&nodeAcc[i], ((float*)nacc)[i]);
}

// ---------------- K6: 64 sr-regulators ----------------
__global__ void k_sr(const float* w1, const float* b1, const float* g, const float* be,
                     const float* w2, const float* b2, float* S,
                     const float* nodeAcc, float* infl, float* dout_ent)
{
    int n = threadIdx.x;
    if (n < 64) {
        float M2 = (float)BB * (float)DD;
        float var_n = (nodeAcc[64 + n] - nodeAcc[n] * nodeAcc[n] / M2) / (M2 - 1.f);
        float ent_n = nodeAcc[128 + n] / (float)BB;
        float sig[3] = {var_n, ent_n, S[S_ORTHO]};
        float out[3];
        regulator3(sig, 3, w1, b1, g, be, w2, b2, out);
        infl[n] = out[0];
        if (n == 63) { S[S_ENT_LAST] = ent_n; *dout_ent = ent_n; }
    }
}

// ---------------- K7: apply symbiotic blend + flat stats ----------------
__global__ void __launch_bounds__(256) k_sym(const float* __restrict__ xp,
                                             const float* __restrict__ Wq,
                                             const float* __restrict__ basis,
                                             const float* __restrict__ infl,
                                             float* __restrict__ xo, float* S)
{
    __shared__ __align__(16) float in_s[8][1024];
    __shared__ float wq_s[256];
    __shared__ float infl_s[64];
    __shared__ float red[256];
    int t = threadIdx.x, lane = t & 31, wave = t >> 5;
    for (int i = t; i < 256; i += 256) wq_s[i] = Wq[i];
    for (int i = t; i < 64; i += 256) infl_s[i] = infl[i];
    float b0[16], b1v[16];
    for (int d = 0; d < 16; ++d) { b0[d] = basis[d]; b1v[d] = basis[16 + d]; }
    const float scale = 4.0f + 1e-8f;
    float acc_s = 0.f, acc_q = 0.f;
    for (int g = blockIdx.x; g < BB / 8; g += gridDim.x) {
        __syncthreads();
        const float4* src = (const float4*)(xp + (size_t)g * 8 * HID);
        for (int i = t; i < 2048; i += 256)
            async_copy_b128(&((float4*)in_s)[i], &src[i]);
        async_wait0();
        __syncthreads();
        float* row = in_s[wave];
        float* dst = xo + (size_t)(g * 8 + wave) * HID;
        for (int nn = 0; nn < 2; ++nn) {
            int n = lane + nn * 32;
            float xv[16];
            for (int d = 0; d < 16; ++d) xv[d] = row[n * 16 + d];
            float q[16];
            for (int dp = 0; dp < 16; ++dp) {
                float v = 0.f;
                for (int d = 0; d < 16; ++d) v += wq_s[dp * 16 + d] * xv[d];
                q[dp] = v;
            }
            float l0 = 0.f, l1 = 0.f;
            for (int d = 0; d < 16; ++d) { l0 += b0[d] * q[d]; l1 += b1v[d] * q[d]; }
            l0 /= scale; l1 /= scale;
            float m = fmaxf(l0, l1);
            float e0 = expf(l0 - m), e1 = expf(l1 - m);
            float inv = 1.f / (e0 + e1);
            float w0 = e0 * inv, w1 = e1 * inv;
            float fi = infl_s[n];
            for (int d = 0; d < 16; ++d) {
                float r = (1.f - fi) * xv[d] + fi * (w0 * b0[d] + w1 * b1v[d]);
                dst[n * 16 + d] = r;
                acc_s += r; acc_q += r * r;
            }
        }
    }
    __syncthreads();
    red[t] = acc_s; __syncthreads();
    for (int st = 128; st > 0; st >>= 1) { if (t < st) red[t] += red[t + st]; __syncthreads(); }
    if (t == 0) atomicAdd(&S[S_SUM_FL], red[0]);
    __syncthreads(); red[t] = acc_q; __syncthreads();
    for (int st = 128; st > 0; st >>= 1) { if (t < st) red[t] += red[t + st]; __syncthreads(); }
    if (t == 0) atomicAdd(&S[S_SUMSQ_FL], red[0]);
}

// ---------------- K8: scr-regulator (1 thread) ----------------
__global__ void k_scr(const float* w1, const float* b1, const float* g, const float* be,
                      const float* w2, const float* b2, float* S)
{
    float M = (float)BB * (float)HID;
    float varfl = (S[S_SUMSQ_FL] - S[S_SUM_FL] * S[S_SUM_FL] / M) / (M - 1.f);
    float sig[2] = {varfl, S[S_ENT_LAST]};
    float out[3];
    regulator3(sig, 2, w1, b1, g, be, w2, b2, out);
    S[S_GAIN] = out[0];
}

// ---------------- K9: readout + SupCon proj (WMMA, two N-tiles, K=1024) ----------------
__global__ void __launch_bounds__(256) k_read(const float* __restrict__ xf,
                                              const float* __restrict__ Wr,
                                              const float* __restrict__ br,
                                              const float* __restrict__ sw1,
                                              const float* __restrict__ sw2,
                                              const float* S,
                                              float* __restrict__ out_logits,
                                              float* __restrict__ out_proj)
{
    __shared__ float hbuf[8][16][8];
    int t = threadIdx.x, lane = t & 31, wave = t >> 5;
    int tile = blockIdx.x * 8 + wave;      // 2048 batch tiles
    int row = lane & 15, half = lane >> 4;
    int col = row;
    const float* arow = xf + (size_t)(tile * 16 + row) * HID;
    const float* wrow = (col < NC) ? (Wr  + (size_t)col * HID) : nullptr;
    const float* srow = (col < 8)  ? (sw1 + (size_t)col * HID) : nullptr;
    if (wrow) __builtin_prefetch(wrow, 0, 3);
    if (srow) __builtin_prefetch(srow, 0, 3);

    v8f cr_ = {0.f, 0.f, 0.f, 0.f, 0.f, 0.f, 0.f, 0.f};
    v8f cs_ = {0.f, 0.f, 0.f, 0.f, 0.f, 0.f, 0.f, 0.f};
    for (int s = 0; s < 32; ++s) {
        int kg = s * 32;
        v16h a, bw, bs;
        int ka = kg + half * 16;
        for (int u = 0; u < 8; ++u) {
            a[u]     = (_Float16)arow[ka + u];
            a[u + 8] = (_Float16)arow[ka + 8 + u];
        }
        int kb = kg + half * 16;
        for (int u = 0; u < 16; ++u) {
            bw[u] = wrow ? (_Float16)wrow[kb + u] : (_Float16)0.f;
            bs[u] = srow ? (_Float16)srow[kb + u] : (_Float16)0.f;
        }
        cr_ = __builtin_amdgcn_wmma_f32_16x16x32_f16(false, a, false, bw, (short)0, cr_, false, false);
        cs_ = __builtin_amdgcn_wmma_f32_16x16x32_f16(false, a, false, bs, (short)0, cs_, false, false);
    }
    int rowoff = half * 8;
    float bias = (col < NC) ? br[col] : 0.f;
    for (int r = 0; r < 8; ++r) {
        int grow = tile * 16 + rowoff + r;
        if (col < NC) out_logits[(size_t)grow * NC + col] = cr_[r] + bias;
        float h = fmaxf(cs_[r], 0.f);
        if (col < 8) hbuf[wave][rowoff + r][col] = h;
    }
    __syncthreads();
    float gain = S[S_GAIN];
    for (int k2 = 0; k2 < 2; ++k2) {
        int idx = lane + k2 * 32;   // 0..63 -> 16 rows x 4 proj channels
        int r = idx >> 2, p = idx & 3;
        float s = 0.f;
        for (int h = 0; h < 8; ++h) s += hbuf[wave][r][h] * sw2[p * 8 + h];
        out_proj[(size_t)(tile * 16 + r) * NPROJ + p] = s * gain;
    }
}

extern "C" void kernel_launch(void* const* d_in, const int* in_sizes, int n_in,
                              void* d_out, int out_size, void* d_ws, size_t ws_size,
                              hipStream_t stream)
{
    const float* x     = (const float*)d_in[0];
    const float* We    = (const float*)d_in[1];
    const float* be_   = (const float*)d_in[2];
    const float* adjw  = (const float*)d_in[3];
    const float* mask  = (const float*)d_in[4];
    const float* Vs    = (const float*)d_in[5];
    const float* Ws    = (const float*)d_in[6];
    const float* gw    = (const float*)d_in[7];
    const float* gb    = (const float*)d_in[8];
    const float* Wf    = (const float*)d_in[9];
    const float* basis = (const float*)d_in[10];
    const float* Wq    = (const float*)d_in[11];
    const float* scw1  = (const float*)d_in[12];
    const float* scw2  = (const float*)d_in[13];
    const float* Wr    = (const float*)d_in[14];
    const float* br    = (const float*)d_in[15];
    const float* pr_[6]; const float* cr_[6]; const float* sr_[6]; const float* scr_[6];
    for (int i = 0; i < 6; ++i) {
        pr_[i]  = (const float*)d_in[16 + i];
        cr_[i]  = (const float*)d_in[22 + i];
        sr_[i]  = (const float*)d_in[28 + i];
        scr_[i] = (const float*)d_in[34 + i];
    }

    float* ws      = (float*)d_ws;
    float* buf0    = ws;                       // 33554432 floats (x_embed -> x_proc)
    float* buf1    = ws + 33554432ull;         // 33554432 floats (x_agg  -> refined)
    float* S       = ws + 67108864ull;         // scalars
    float* nodeAcc = S + 512;                  // 192
    float* infl    = nodeAcc + 192;            // 64
    float* adjm    = infl + 64;                // 4096
    float* out     = (float*)d_out;
    float* out_logits = out;                       // 327680
    float* out_proj   = out + (size_t)BB * NC;     // 131072
    float* out_ent    = out + (size_t)BB * NC + (size_t)BB * NPROJ;      // [458752]
    float* out_ortho  = out_ent + 1;                                      // [458753]

    k_zero  <<<1, 256, 0, stream>>>(S);
    k_xstats<<<256, 256, 0, stream>>>(x, S, BB * FIN);
    k_embed <<<4096, 256, 0, stream>>>(x, We, be_, buf0, S);
    k_prep  <<<1, 256, 0, stream>>>(adjw, mask, Ws, basis,
                                    pr_[0], pr_[1], pr_[2], pr_[3], pr_[4], pr_[5],
                                    S, adjm, out_ortho);
    k_agg   <<<1024, 256, 0, stream>>>(buf0, adjm, Vs, buf1, S);
    k_cr    <<<1, 1, 0, stream>>>(cr_[0], cr_[1], cr_[2], cr_[3], cr_[4], cr_[5], S);
    k_cont  <<<1024, 256, 0, stream>>>(buf1, Vs, Wf, Wq, gw, gb, basis, S, nodeAcc, buf0);
    k_sr    <<<1, 64, 0, stream>>>(sr_[0], sr_[1], sr_[2], sr_[3], sr_[4], sr_[5],
                                   S, nodeAcc, infl, out_ent);
    k_sym   <<<1024, 256, 0, stream>>>(buf0, Wq, basis, infl, buf1, S);
    k_scr   <<<1, 1, 0, stream>>>(scr_[0], scr_[1], scr_[2], scr_[3], scr_[4], scr_[5], S);
    k_read  <<<256, 256, 0, stream>>>(buf1, Wr, br, scw1, scw2, S, out_logits, out_proj);
}
